// TreeRecursiveNN_56951266345607
// MI455X (gfx1250) — matrix operand order repaired and checked by reading, not compiled, not run on hardware
//
#include <hip/hip_runtime.h>
#include <hip/hip_bf16.h>

// ---------------------------------------------------------------------------
// Tree-LSTM for MI455X (gfx1250, wave32, WMMA bf16 16x16x32, f32 accum).
//
//   prep_xg     : xg_table[v][g][j] = emb[v] @ W[g] + bW[g]        (64x4x256 f32)
//   prep_upack  : U -> Ucat(512 x 1280) packed in per-lane B-fragment order (bf16)
//   leaf_kernel : level 10 cell (children = 0); writes c (f32) + h packed-bf16
//   level_kernel: lev 9..1: fused GEMM (M x 512)x(512 x 1280) + LSTM cell
//                 10 waves/block, exactly one 16x16 WMMA tile per wave ->
//                 branch-free inner loop, EXEC all-ones for every WMMA.
//   scores_kern : root_hidden @ Ws + bs
//
// WMMA wave32 fragment maps (ISA 7.12.2):
//   A 16x32 bf16: lane l -> m = l&15, grp = l>>4; elem e -> k = e + 8*((e>>3)+grp)
//   B 32x16 bf16: lane l -> n = l&15, grp = l>>4; elem e -> k = e + 8*((e>>3)+grp)
//   C 16x16 f32 : lane l -> n = l&15; vgpr r -> m = r + 8*(l>>4)
// ---------------------------------------------------------------------------

typedef __attribute__((ext_vector_type(16))) __bf16 v16bf;
typedef __attribute__((ext_vector_type(8)))  float  v8f;

#define Bsz   64
#define TOTAL 2046

// workspace layout (bytes)
#define XG_OFF   ((size_t)0)                       // 64*4*256 f32      = 256 KB
#define UP_OFF   ((size_t)262144)                  // 512*1280 bf16     = 1.25 MB
#define H0_OFF   ((size_t)1572864)                 // 32 MB slot
#define H1_OFF   (H0_OFF + (size_t)33554432)       // 32 MB slot
#define C0_OFF   (H1_OFF + (size_t)33554432)       // 64 MB slot
#define C1_OFF   (C0_OFF + (size_t)67108864)       // 64 MB slot

__device__ __forceinline__ float sigf(float x) { return 1.0f / (1.0f + __expf(-x)); }

// store one h value into packed A-fragment layout for the NEXT level's GEMM,
// which views this h buffer as (M/2, 512).  pflat = m_child*256 + j.
__device__ __forceinline__ void store_h_packed(__hip_bfloat16* hp, int pflat, float v) {
    int mA  = pflat >> 9;           // row in (M/2, 512) view
    int kA  = pflat & 511;
    int mt  = mA >> 4, rA = mA & 15;
    int kt  = kA >> 5, kk = kA & 31;
    int grp = (kk >> 3) & 1;
    int e   = (kk & 7) | ((kk >> 4) << 3);
    int lane = rA | (grp << 4);
    hp[((((mt << 4) + kt) * 32 + lane) << 4) + e] = __float2bfloat16(v);
}

__device__ __forceinline__ v16bf load_frag(const __hip_bfloat16* base, int tile, int lane) {
    return *reinterpret_cast<const v16bf*>(base + (((size_t)tile * 32 + lane) << 4));
}

// ---------------------------------------------------------------------------
__global__ __launch_bounds__(256) void prep_xg(const float* __restrict__ emb,
                                               const float* __restrict__ W,
                                               const float* __restrict__ bW,
                                               float* __restrict__ xg) {
    int idx = blockIdx.x * 256 + threadIdx.x;          // 65536 outputs
    int v = idx >> 10, g = (idx >> 8) & 3, j = idx & 255;
    float acc = bW[g * 256 + j];
    for (int e = 0; e < 256; ++e)
        acc += emb[v * 256 + e] * W[(g * 256 + e) * 256 + j];
    xg[idx] = acc;
}

// U(5,2,256,256) f32 -> Ucat(512,1280) bf16 packed in B-fragment order.
__global__ __launch_bounds__(256) void prep_upack(const float* __restrict__ U,
                                                  __hip_bfloat16* __restrict__ up) {
    int idx  = blockIdx.x * 256 + threadIdx.x;         // 655360 halves
    int e    = idx & 15;
    int lane = (idx >> 4) & 31;
    int tile = idx >> 9;
    int kt   = tile & 15;                              // K32 block (0..15)
    int nt   = tile >> 4;                              // N16 block (0..79)
    int grp  = lane >> 4;
    int ncol = nt * 16 + (lane & 15);                  // 0..1279
    int g    = ncol >> 8;                              // gate 0..4
    int j    = ncol & 255;
    int kcat = (kt << 5) + (e & 7) + (grp << 3) + ((e >> 3) << 4); // 0..511
    int ch   = kcat >> 8;                              // left/right child
    int kr   = kcat & 255;
    up[idx] = __float2bfloat16(U[((g * 2 + ch) * 256 + kr) * 256 + j]);
}

// ---------------------------------------------------------------------------
// level 10 (leaves): children are zero -> c = i*u, h = o*tanh(c)
__global__ __launch_bounds__(256) void leaf_kernel(const int* __restrict__ node_ids,
                                                   const float* __restrict__ xg,
                                                   float* __restrict__ cout,
                                                   __hip_bfloat16* __restrict__ hout) {
    int idx = blockIdx.x * 256 + threadIdx.x;          // 64*1024*256 threads
    int j  = idx & 255;
    int mc = idx >> 8;                                 // b*1024 + node
    int b = mc >> 10, node = mc & 1023;
    int id = node_ids[b * TOTAL + 1022 + node];
    const float* xt = xg + id * 1024;
    float ig = sigf(xt[256 + j]);
    float og = sigf(xt[512 + j]);
    float ug = tanhf(xt[768 + j]);
    float c  = ig * ug;
    cout[idx] = c;
    store_h_packed(hout, idx, og * tanhf(c));
}

// ---------------------------------------------------------------------------
// Fused GEMM + cell for levels 9..1.
// Block: 320 thr = 10 waves. Tile: 32 rows x 16 j-cols x 5 gates = 10 WMMA
// tiles; wave w owns tile (mt = w&1, gate = w>>1). No branches in the k-loop.
__global__ __launch_bounds__(320) void level_kernel(
    const __hip_bfloat16* __restrict__ hin,  // child h, packed A frags, (M,512) view
    const float* __restrict__ cin,           // child c, linear f32
    const int* __restrict__ node_ids,
    const float* __restrict__ xg,
    const __hip_bfloat16* __restrict__ up,   // packed Ucat B frags
    float* __restrict__ cout,
    __hip_bfloat16* __restrict__ hout,
    float* __restrict__ root_out,            // non-null only at lev==1
    int n, int lev) {
    __shared__ float lds[5 * 32 * 16];

    int tid  = threadIdx.x;
    int wave = tid >> 5, lane = tid & 31;
    int mt_loc  = wave & 1;
    int g       = wave >> 1;                           // 0..4
    int mt_glob = blockIdx.x * 2 + mt_loc;
    int ntB     = g * 16 + blockIdx.y;                 // gate g, j-block

    v8f acc = {};
    for (int kt = 0; kt < 16; ++kt) {                  // K = 512
        v16bf a = load_frag(hin, mt_glob * 16 + kt, lane);
        v16bf b = load_frag(up, ntB * 16 + kt, lane);
        acc = __builtin_amdgcn_wmma_f32_16x16x32_bf16(false, a, false, b,
                                                      (short)0, acc, false, false);
    }

    int row0 = (mt_loc << 4) + ((lane >> 4) << 3);     // C layout: m = r + 8*grp
    int col  = lane & 15;
#pragma unroll
    for (int r = 0; r < 8; ++r)
        lds[(g * 32 + row0 + r) * 16 + col] = acc[r];
    __syncthreads();

    // epilogue: 32x16 cells over 320 threads
    for (int idx = tid; idx < 512; idx += 320) {
        int r = idx >> 4, jj = idx & 15;
        int m = blockIdx.x * 32 + r;                   // m = b*n + node
        int j = (blockIdx.y << 4) + jj;
        int b    = m >> lev;
        int node = m & (n - 1);
        int id   = node_ids[b * TOTAL + (n - 2) + node];
        const float* xt = xg + id * 1024;
        float p0 = lds[(0 * 32 + r) * 16 + jj] + xt[j];          // f_l (gate_map 0)
        float p1 = lds[(1 * 32 + r) * 16 + jj] + xt[j];          // f_r (gate_map 0)
        float p2 = lds[(2 * 32 + r) * 16 + jj] + xt[256 + j];    // i
        float p3 = lds[(3 * 32 + r) * 16 + jj] + xt[512 + j];    // o
        float p4 = lds[(4 * 32 + r) * 16 + jj] + xt[768 + j];    // u
        float fl = sigf(p0), fr = sigf(p1), ig = sigf(p2), og = sigf(p3);
        float ug = tanhf(p4);
        float cl = cin[(2 * m) * 256 + j];                       // children 2m,2m+1
        float cr = cin[(2 * m + 1) * 256 + j];
        float cn = ig * ug + fl * cl + fr * cr;
        cout[m * 256 + j] = cn;
        float hv = og * tanhf(cn);
        if (root_out) root_out[m * 256 + j] = hv;                // lev 1: root_hidden
        else          store_h_packed(hout, m * 256 + j, hv);
    }
}

// ---------------------------------------------------------------------------
__global__ void scores_kernel(const float* __restrict__ rh,
                              const float* __restrict__ Wsp,
                              const float* __restrict__ bsp,
                              float* __restrict__ out) {
    int tid = threadIdx.x;
    if (tid >= 192) return;
    int b = tid / 3, s = tid % 3;
    float acc = bsp[s];
    for (int k = 0; k < 512; ++k)
        acc += rh[b * 512 + k] * Wsp[k * 3 + s];
    out[b * 3 + s] = acc;
}

// ---------------------------------------------------------------------------
extern "C" void kernel_launch(void* const* d_in, const int* in_sizes, int n_in,
                              void* d_out, int out_size, void* d_ws, size_t ws_size,
                              hipStream_t stream) {
    (void)in_sizes; (void)n_in; (void)out_size; (void)ws_size;
    const int*   node_ids = (const int*)d_in[0];
    const float* emb      = (const float*)d_in[1];
    const float* W        = (const float*)d_in[2];
    const float* bW       = (const float*)d_in[3];
    const float* U        = (const float*)d_in[4];
    const float* Ws       = (const float*)d_in[5];
    const float* bs       = (const float*)d_in[6];
    float* out = (float*)d_out;                 // [0,192) scores, [192,32960) root_hidden
    char*  ws  = (char*)d_ws;

    float*          xg = (float*)(ws + XG_OFF);
    __hip_bfloat16* up = (__hip_bfloat16*)(ws + UP_OFF);
    __hip_bfloat16* h0 = (__hip_bfloat16*)(ws + H0_OFF);
    __hip_bfloat16* h1 = (__hip_bfloat16*)(ws + H1_OFF);
    float*          c0 = (float*)(ws + C0_OFF);
    float*          c1 = (float*)(ws + C1_OFF);

    prep_xg   <<<256,   256, 0, stream>>>(emb, W, bW, xg);
    prep_upack<<<2560,  256, 0, stream>>>(U, up);
    leaf_kernel<<<65536, 256, 0, stream>>>(node_ids, xg, c0, h0);

    __hip_bfloat16* hin = h0; __hip_bfloat16* hout = h1;
    float* cin = c0; float* cout = c1;
    for (int lev = 9; lev >= 1; --lev) {
        int n = 1 << lev;
        int M = Bsz * n;
        dim3 grid(M / 32, 16);
        level_kernel<<<grid, 320, 0, stream>>>(hin, cin, node_ids, xg, up,
                                               cout, hout,
                                               (lev == 1) ? (out + 192) : nullptr,
                                               n, lev);
        __hip_bfloat16* ht = hin; hin = hout; hout = ht;
        float* ct = cin; cin = cout; cout = ct;
    }

    scores_kernel<<<1, 192, 0, stream>>>(out + 192, Ws, bs, out);
}
